// GlobalMoEBlock_60163901883068
// MI455X (gfx1250) — compile-verified, matrix operand order checked by reading, not run on hardware
//
#include <hip/hip_runtime.h>
#include <hip/hip_bf16.h>

// ---------------------------------------------------------------------------
// MoE block for MI455X (gfx1250): router + top-8 gather + per-expert bf16 WMMA.
// Fast path: one-time fp32->bf16 transpose of weights; M=32 token tiles so
// every streamed B fragment feeds two WMMAs (halved weight traffic per FLOP).
// ---------------------------------------------------------------------------

typedef __attribute__((ext_vector_type(16))) __bf16 v16bf;
typedef __attribute__((ext_vector_type(8)))  __bf16 v8bf;
typedef __attribute__((ext_vector_type(8)))  float  v8f;

namespace {
constexpr int kH = 1024;   // hidden
constexpr int kE = 64;     // experts
constexpr int kI = 512;    // intermediate
constexpr int kN = 2048;   // tokens (B*T)
constexpr int kTopK = 8;
constexpr int kCap = kN;   // per-expert list capacity

constexpr int kAStride = kH + 8;   // LDS row stride (ushorts); 2064B % 16 == 0
constexpr int kCStride = kI + 8;   // LDS row stride (ushorts); 1040B % 16 == 0

// Workspace layout
constexpr size_t kOffCounts = 0;
constexpr size_t kOffTok    = 256;
constexpr size_t kOffGate   = kOffTok + (size_t)kE * kCap * 4;
constexpr size_t kOffXbf    = kOffGate + (size_t)kE * kCap * 4;
constexpr size_t kOffWgT    = kOffXbf + (size_t)kN * kH * 2;
constexpr size_t kOffWuT    = kOffWgT + (size_t)kE * kH * kI * 2;
constexpr size_t kOffWdT    = kOffWuT + (size_t)kE * kH * kI * 2;
constexpr size_t kWsFast    = kOffWdT + (size_t)kE * kI * kH * 2;
}

__device__ __forceinline__ unsigned short f32_to_bf16_rne(float f) {
  unsigned u = __builtin_bit_cast(unsigned, f);
  unsigned r = u + 0x7FFFu + ((u >> 16) & 1u);
  return (unsigned short)(r >> 16);
}

__device__ __forceinline__ __bf16 us_as_bf16(unsigned short s) {
  return __builtin_bit_cast(__bf16, s);
}

// K offset within a 32-chunk for packed pair j (CDNA5 16-bit A/B layout).
__device__ __forceinline__ int wmma_k0(int j) {
  return ((j >> 2) << 4) + ((j & 3) << 1);
}

// Per-lane fragment base for row-major bf16 [R][rowStride]: row = lane&15,
// K base = 8*(lane>>4). Fragment = two contiguous 16B chunks at +kk, +kk+16.
__device__ __forceinline__ const unsigned short* frag_base(
    const unsigned short* base, int rowStride) {
  const int lane = threadIdx.x & 31;
  return base + (size_t)(lane & 15) * rowStride + ((lane >> 4) << 3);
}

__device__ __forceinline__ v16bf load_frag(const unsigned short* laneBase,
                                           int kk) {
  v8bf lo = *(const v8bf*)(laneBase + kk);
  v8bf hi = *(const v8bf*)(laneBase + kk + 16);
  return __builtin_shufflevector(lo, hi, 0, 1, 2, 3, 4, 5, 6, 7, 8, 9, 10, 11,
                                 12, 13, 14, 15);
}

// Fallback helpers (small-workspace path): convert fp32 weights in registers.
__device__ __forceinline__ v16bf load_frag_bf16(const unsigned short* base,
                                                int rowStride, int kk) {
  return load_frag(frag_base(base, rowStride), kk);
}

__device__ __forceinline__ v16bf load_b_frag_f32(const float* __restrict__ W,
                                                 int ld) {
  const int lane = threadIdx.x & 31;
  const int n = lane & 15;
  const int kb = (lane >> 4) << 3;
  const float* p = W + (size_t)kb * ld + n;
  v16bf b;
#pragma unroll
  for (int j = 0; j < 8; ++j) {
    const int k0 = wmma_k0(j);
    b[2 * j]     = us_as_bf16(f32_to_bf16_rne(p[(size_t)k0 * ld]));
    b[2 * j + 1] = us_as_bf16(f32_to_bf16_rne(p[(size_t)(k0 + 1) * ld]));
  }
  return b;
}

__device__ __forceinline__ v8f wmma_bf16(v16bf a, v16bf b, v8f c) {
  return __builtin_amdgcn_wmma_f32_16x16x32_bf16(
      /*neg_a=*/false, a, /*neg_b=*/false, b,
      /*c_mod=*/(short)0, c, /*reuse_a=*/false, /*reuse_b=*/false);
}

__device__ __forceinline__ float fast_silu(float g) {
  return g * __builtin_amdgcn_rcpf(1.0f + __expf(-g));
}

// ---------------------------------------------------------------------------
// Kernel 0: zero combine-output region and per-expert counters.
// ---------------------------------------------------------------------------
__global__ void moe_init(float* __restrict__ out, int* __restrict__ counts,
                         size_t total) {
  size_t i = (size_t)blockIdx.x * blockDim.x + threadIdx.x;
  if (i < total) out[i] = 0.0f;
  if (i < kE) counts[i] = 0;
}

// ---------------------------------------------------------------------------
// Kernel 0b: x fp32 -> bf16 (same layout).
// ---------------------------------------------------------------------------
__global__ void convert_x(const float* __restrict__ src,
                          unsigned short* __restrict__ dst, size_t total) {
  size_t i = (size_t)blockIdx.x * blockDim.x + threadIdx.x;
  if (i < total) dst[i] = f32_to_bf16_rne(src[i]);
}

// ---------------------------------------------------------------------------
// Kernel 0c: per-expert transpose + convert: src [E][K][N] f32 -> dst [E][N][K]
// bf16. 32x32 tiles through LDS; reads and writes both coalesced.
// ---------------------------------------------------------------------------
__global__ __launch_bounds__(256) void convert_transpose(
    const float* __restrict__ src, unsigned short* __restrict__ dst, int K,
    int N) {
  __shared__ float tile[32][33];
  const int e = blockIdx.z;
  const size_t eo = (size_t)e * K * N;
  const int n0 = blockIdx.x * 32;
  const int k0 = blockIdx.y * 32;
  const int tx = threadIdx.x & 31;
  const int ty = threadIdx.x >> 5;  // 0..7
  for (int r = ty; r < 32; r += 8)
    tile[r][tx] = src[eo + (size_t)(k0 + r) * N + (n0 + tx)];
  __syncthreads();
  for (int r = ty; r < 32; r += 8)
    dst[eo + (size_t)(n0 + r) * K + (k0 + tx)] = f32_to_bf16_rne(tile[tx][r]);
}

// ---------------------------------------------------------------------------
// Kernel 1: router. One 64-thread block per token.
// ---------------------------------------------------------------------------
__global__ __launch_bounds__(kE) void moe_router(
    const float* __restrict__ x, const float* __restrict__ Wr,
    float* __restrict__ logits_out, int* __restrict__ counts,
    int* __restrict__ list_tok, float* __restrict__ list_gate) {
  const int n = blockIdx.x;
  const int e = threadIdx.x;  // 0..63

  __shared__ float sx[kH];
  __shared__ float sp[kE];
  __shared__ float sred[2];

  for (int h = e; h < kH; h += kE) sx[h] = x[(size_t)n * kH + h];
  __syncthreads();

  float acc = 0.0f;
  for (int h = 0; h < kH; ++h) acc += sx[h] * Wr[h * kE + e];
  logits_out[(size_t)n * kE + e] = acc;

  sp[e] = acc;
  __syncthreads();
  if (e == 0) {
    float m = sp[0];
    for (int i = 1; i < kE; ++i) m = fmaxf(m, sp[i]);
    sred[0] = m;
  }
  __syncthreads();
  float p = expf(acc - sred[0]);
  sp[e] = p;
  __syncthreads();
  if (e == 0) {
    float s = 0.0f;
    for (int i = 0; i < kE; ++i) s += sp[i];
    sred[1] = s;
  }
  __syncthreads();
  sp[e] = p / sred[1];
  __syncthreads();

  if (e == 0) {
    int ti[kTopK];
    float tv[kTopK];
    float vsum = 0.0f;
    for (int k = 0; k < kTopK; ++k) {
      int best = 0;
      float bv = sp[0];
      for (int i = 1; i < kE; ++i) {
        if (sp[i] > bv) { bv = sp[i]; best = i; }  // first max, like top_k
      }
      ti[k] = best;
      tv[k] = bv;
      sp[best] = -1.0e30f;
      vsum += bv;
    }
    for (int k = 0; k < kTopK; ++k) {
      const int ex = ti[k];
      const float gate = tv[k] / vsum;
      const int pos = atomicAdd(&counts[ex], 1);
      list_tok[ex * kCap + pos] = n;
      list_gate[ex * kCap + pos] = gate;
    }
  }
}

// ---------------------------------------------------------------------------
// Kernel 2 (fast): per-expert SwiGLU + combine, bf16 weights pre-transposed.
// grid = (tiles, experts); 32 gathered tokens per tile (2 M-subtiles);
// 256 threads = 8 wave32. Each streamed B fragment feeds 2 WMMAs.
// ---------------------------------------------------------------------------
__global__ __launch_bounds__(256) void moe_expert_fast(
    const unsigned short* __restrict__ xbf,   // [N][H] bf16
    const unsigned short* __restrict__ WgT,   // [E][I][H] bf16
    const unsigned short* __restrict__ WuT,   // [E][I][H] bf16
    const unsigned short* __restrict__ WdT,   // [E][H][I] bf16
    const int* __restrict__ counts, const int* __restrict__ list_tok,
    const float* __restrict__ list_gate, float* __restrict__ out) {
  const int e = blockIdx.y;
  const int cnt = counts[e];
  const int r0 = blockIdx.x * 32;
  if (r0 >= cnt) return;  // uniform across block

  __shared__ unsigned short sC[32 * kCStride];  // gated activations, bf16
  __shared__ int sTok[32];
  __shared__ float sGate[32];

  const int tid = threadIdx.x;
  if (tid < 32) {
    const int idx = r0 + tid;
    const bool valid = idx < cnt;
    sTok[tid]  = list_tok[e * kCap + (valid ? idx : r0)];
    sGate[tid] = valid ? list_gate[e * kCap + idx] : 0.0f;  // pad rows -> 0
  }
  __syncthreads();

  const int wave = tid >> 5;
  const int lane = tid & 31;
  const int nIdx = lane & 15;
  const int kb = (lane >> 4) << 3;
  const int mBase = (lane >> 4) << 3;

  // Per-lane A bases: two M-subtiles of gathered bf16 token rows in global.
  const unsigned short* a0base =
      xbf + (size_t)sTok[lane & 15] * kH + kb;
  const unsigned short* a1base =
      xbf + (size_t)sTok[16 + (lane & 15)] * kH + kb;

  const unsigned short* WgE = WgT + (size_t)e * kI * kH;
  const unsigned short* WuE = WuT + (size_t)e * kI * kH;

  // -------- Phase 1: gate/up projections + SwiGLU --------
#pragma unroll
  for (int t = 0; t < 4; ++t) {                 // 8 waves * 4 tiles = I/16
    const int ncol = (wave * 4 + t) * 16;
    const unsigned short* bg0 = frag_base(WgE + (size_t)ncol * kH, kH);
    const unsigned short* bu0 = frag_base(WuE + (size_t)ncol * kH, kH);
    v8f aG0 = {}, aU0 = {}, aG1 = {}, aU1 = {};
#pragma unroll 2
    for (int kk = 0; kk < kH; kk += 32) {
      const v16bf bg = load_frag(bg0, kk);
      const v16bf bu = load_frag(bu0, kk);
      const v16bf a0 = load_frag(a0base, kk);
      const v16bf a1 = load_frag(a1base, kk);
      aG0 = wmma_bf16(a0, bg, aG0);
      aU0 = wmma_bf16(a0, bu, aU0);
      aG1 = wmma_bf16(a1, bg, aG1);
      aU1 = wmma_bf16(a1, bu, aU1);
    }
#pragma unroll
    for (int r = 0; r < 8; ++r) {
      const int m0 = mBase + r;
      const int m1 = 16 + mBase + r;
      const float act0 = fast_silu(aG0[r]) * aU0[r] * sGate[m0];
      const float act1 = fast_silu(aG1[r]) * aU1[r] * sGate[m1];
      sC[m0 * kCStride + ncol + nIdx] = f32_to_bf16_rne(act0);
      sC[m1 * kCStride + ncol + nIdx] = f32_to_bf16_rne(act1);
    }
  }
  __syncthreads();

  // -------- Phase 2: down projection + atomic combine --------
  const unsigned short* WdE = WdT + (size_t)e * kH * kI;
  const unsigned short* c0base = frag_base(sC, kCStride);
  const unsigned short* c1base = c0base + 16 * kCStride;
#pragma unroll
  for (int t = 0; t < 8; ++t) {                 // 8 waves * 8 tiles = H/16
    const int ncol = (wave * 8 + t) * 16;
    const unsigned short* bd0 = frag_base(WdE + (size_t)ncol * kI, kI);
    v8f acc0 = {}, acc1 = {};
#pragma unroll 2
    for (int kk = 0; kk < kI; kk += 32) {
      const v16bf b = load_frag(bd0, kk);
      const v16bf a0 = load_frag(c0base, kk);
      const v16bf a1 = load_frag(c1base, kk);
      acc0 = wmma_bf16(a0, b, acc0);
      acc1 = wmma_bf16(a1, b, acc1);
    }
#pragma unroll
    for (int r = 0; r < 8; ++r) {
      const int m0 = mBase + r;
      const int m1 = 16 + mBase + r;
      unsafeAtomicAdd(&out[(size_t)sTok[m0] * kH + ncol + nIdx], acc0[r]);
      unsafeAtomicAdd(&out[(size_t)sTok[m1] * kH + ncol + nIdx], acc1[r]);
    }
  }
}

// ---------------------------------------------------------------------------
// Kernel 2 (fallback): converts fp32 weights in-loop (small workspace only).
// ---------------------------------------------------------------------------
__global__ __launch_bounds__(256) void moe_expert_direct(
    const float* __restrict__ x, const float* __restrict__ Wg,
    const float* __restrict__ Wu, const float* __restrict__ Wd,
    const int* __restrict__ counts, const int* __restrict__ list_tok,
    const float* __restrict__ list_gate, float* __restrict__ out) {
  const int e = blockIdx.y;
  const int cnt = counts[e];
  const int r0 = blockIdx.x * 16;
  if (r0 >= cnt) return;

  __shared__ unsigned short sA[16 * kAStride];
  __shared__ unsigned short sC[16 * kCStride];
  __shared__ int sTok[16];
  __shared__ float sGate[16];

  const int tid = threadIdx.x;
  if (tid < 16) {
    const int idx = r0 + tid;
    const bool valid = idx < cnt;
    sTok[tid]  = list_tok[e * kCap + (valid ? idx : r0)];
    sGate[tid] = valid ? list_gate[e * kCap + idx] : 0.0f;
  }
  __syncthreads();

  for (int i = tid; i < 16 * kH; i += 256) {
    const int row = i >> 10;
    const int col = i & (kH - 1);
    sA[row * kAStride + col] =
        f32_to_bf16_rne(x[(size_t)sTok[row] * kH + col]);
  }
  __syncthreads();

  const int wave = tid >> 5;
  const int lane = tid & 31;
  const int nIdx = lane & 15;
  const int mBase = (lane >> 4) << 3;

  const float* Wge = Wg + (size_t)e * kH * kI;
  const float* Wue = Wu + (size_t)e * kH * kI;

#pragma unroll
  for (int t = 0; t < 4; ++t) {
    const int ncol = (wave * 4 + t) * 16;
    v8f accG = {};
    v8f accU = {};
    for (int kk = 0; kk < kH; kk += 32) {
      const v16bf a  = load_frag_bf16(sA, kAStride, kk);
      const v16bf bg = load_b_frag_f32(Wge + (size_t)kk * kI + ncol, kI);
      const v16bf bu = load_b_frag_f32(Wue + (size_t)kk * kI + ncol, kI);
      accG = wmma_bf16(a, bg, accG);
      accU = wmma_bf16(a, bu, accU);
    }
#pragma unroll
    for (int r = 0; r < 8; ++r) {
      const int m = mBase + r;
      const float act = fast_silu(accG[r]) * accU[r] * sGate[m];
      sC[m * kCStride + ncol + nIdx] = f32_to_bf16_rne(act);
    }
  }
  __syncthreads();

  const float* Wde = Wd + (size_t)e * kI * kH;
#pragma unroll
  for (int t = 0; t < 8; ++t) {
    const int ncol = (wave * 8 + t) * 16;
    v8f acc = {};
    for (int kk = 0; kk < kI; kk += 32) {
      const v16bf a = load_frag_bf16(sC, kCStride, kk);
      const v16bf b = load_b_frag_f32(Wde + (size_t)kk * kH + ncol, kH);
      acc = wmma_bf16(a, b, acc);
    }
#pragma unroll
    for (int r = 0; r < 8; ++r) {
      const int m = mBase + r;
      unsafeAtomicAdd(&out[(size_t)sTok[m] * kH + ncol + nIdx], acc[r]);
    }
  }
}

// ---------------------------------------------------------------------------
// Host-side launch
// ---------------------------------------------------------------------------
extern "C" void kernel_launch(void* const* d_in, const int* in_sizes, int n_in,
                              void* d_out, int out_size, void* d_ws,
                              size_t ws_size, hipStream_t stream) {
  const float* x  = (const float*)d_in[0];  // [N, H]
  const float* Wr = (const float*)d_in[1];  // [H, E]
  const float* Wg = (const float*)d_in[2];  // [E, H, I]
  const float* Wu = (const float*)d_in[3];  // [E, H, I]
  const float* Wd = (const float*)d_in[4];  // [E, I, H]

  float* out    = (float*)d_out;            // [N, H]
  float* logits = out + (size_t)kN * kH;    // [N, E]

  char* ws = (char*)d_ws;
  int*   counts    = (int*)(ws + kOffCounts);
  int*   list_tok  = (int*)(ws + kOffTok);
  float* list_gate = (float*)(ws + kOffGate);

  const size_t totalOut = (size_t)kN * kH;
  moe_init<<<(unsigned)((totalOut + 255) / 256), 256, 0, stream>>>(out, counts,
                                                                   totalOut);
  moe_router<<<kN, kE, 0, stream>>>(x, Wr, logits, counts, list_tok,
                                    list_gate);

  if (ws_size >= kWsFast) {
    unsigned short* xbf = (unsigned short*)(ws + kOffXbf);
    unsigned short* WgT = (unsigned short*)(ws + kOffWgT);
    unsigned short* WuT = (unsigned short*)(ws + kOffWuT);
    unsigned short* WdT = (unsigned short*)(ws + kOffWdT);

    const size_t totalX = (size_t)kN * kH;
    convert_x<<<(unsigned)((totalX + 255) / 256), 256, 0, stream>>>(x, xbf,
                                                                    totalX);
    {
      dim3 g(kI / 32, kH / 32, kE);  // K=H, N=I
      convert_transpose<<<g, 256, 0, stream>>>(Wg, WgT, kH, kI);
      convert_transpose<<<g, 256, 0, stream>>>(Wu, WuT, kH, kI);
    }
    {
      dim3 g(kH / 32, kI / 32, kE);  // K=I, N=H
      convert_transpose<<<g, 256, 0, stream>>>(Wd, WdT, kI, kH);
    }
    dim3 grid(kN / 32, kE);  // tiles fastest-varying -> same-expert L2 reuse
    moe_expert_fast<<<grid, 256, 0, stream>>>(xbf, WgT, WuT, WdT, counts,
                                              list_tok, list_gate, out);
  } else {
    dim3 grid(kN / 16, kE);
    moe_expert_direct<<<grid, 256, 0, stream>>>(x, Wg, Wu, Wd, counts,
                                                list_tok, list_gate, out);
  }
}
